// Model_45973329936794
// MI455X (gfx1250) — matrix operand order repaired and checked by reading, not compiled, not run on hardware
//
#include <hip/hip_runtime.h>
#include <cstdint>
#include <cstddef>

// Problem constants (match the reference).
#define NT 200000
#define NP 5000
#define FDIM 384
#define CDIM 128
#define ECNT 600000
#define ELAB 400000

typedef __attribute__((ext_vector_type(16))) __bf16 v16bf;
typedef __attribute__((ext_vector_type(8)))  __bf16 v8bf;
typedef __attribute__((ext_vector_type(4)))  __bf16 v4bf;
typedef __attribute__((ext_vector_type(8)))  float  v8f;
typedef __attribute__((ext_vector_type(4)))  float  v4f;

__device__ __forceinline__ int imin(int a, int b) { return a < b ? a : b; }

__device__ __forceinline__ v8f wmma_bf16(v16bf a, v16bf b, v8f c) {
  // D = A(16x32 bf16) * B(32x16 bf16) + C(16x16 f32)
  return __builtin_amdgcn_wmma_f32_16x16x32_bf16(false, a, false, b, (short)0, c,
                                                 false, false);
}

// ---- A-fragment loaders -----------------------------------------------------
// 16-bit A 16x32 layout (ISA 7.12.2): lanes 0-15 hold row M=lane, K = 0..7 and
// 16..23; lanes 16-31 hold row M=lane-16, K = 8..15 and 24..31.  Two contiguous
// 16-byte chunks per lane -> two global_load_b128 per fragment.
__device__ __forceinline__ v16bf load_a_bf16(const __bf16* base, int row, int kk,
                                             int lane) {
  const __bf16* p = base + (size_t)row * CDIM + kk + ((lane >> 4) << 3);
  v8bf lo = *reinterpret_cast<const v8bf*>(p);
  v8bf hi = *reinterpret_cast<const v8bf*>(p + 16);
  v16bf a;
#pragma unroll
  for (int i = 0; i < 8; ++i) { a[i] = lo[i]; a[i + 8] = hi[i]; }
  return a;
}

// Same fragment from an f32 source, converted in-register (v_cvt_pk_bf16_f32).
// Used by the encoder (reads thesis_x directly) and by the layer GEMM for the
// f32 aggregation accumulators (avoids a 200MB convert round-trip).
__device__ __forceinline__ v16bf load_a_f32(const float* base, int ld, int row,
                                            int kk, int lane) {
  const float* p = base + (size_t)row * ld + kk + ((lane >> 4) << 3);
  v4f f0 = *reinterpret_cast<const v4f*>(p);
  v4f f1 = *reinterpret_cast<const v4f*>(p + 4);
  v4f f2 = *reinterpret_cast<const v4f*>(p + 16);
  v4f f3 = *reinterpret_cast<const v4f*>(p + 20);
  v16bf a;
#pragma unroll
  for (int i = 0; i < 4; ++i) {
    a[i]      = (__bf16)f0[i];
    a[i + 4]  = (__bf16)f1[i];
    a[i + 8]  = (__bf16)f2[i];
    a[i + 12] = (__bf16)f3[i];
  }
  return a;
}

__device__ __forceinline__ v16bf load_a_any(const void* src, int isF32, int ld,
                                            int row, int kk, int lane) {
  return isF32 ? load_a_f32((const float*)src, ld, row, kk, lane)
               : load_a_bf16((const __bf16*)src, row, kk, lane);
}

// ---- B fragments ------------------------------------------------------------
// Weights pre-packed so lane l's 16 bf16 B values for tile (kt,nt) are
// contiguous: pack[(((kt*ntiles)+nt)*32 + l)*16 + i].  Lane layout per ISA:
// lanes 0-15 -> col N=lane, K = kt*32+0..7,16..23 ; lanes 16-31 -> K = kt*32+8..15,24..31.
__device__ __forceinline__ v16bf load_b_pack(const __bf16* pack, int kt, int nt,
                                             int ntiles, int lane) {
  const __bf16* p = pack + ((((size_t)kt * ntiles + nt) * 32) + lane) * 16;
  v8bf lo = *reinterpret_cast<const v8bf*>(p);
  v8bf hi = *reinterpret_cast<const v8bf*>(p + 8);
  v16bf b;
#pragma unroll
  for (int i = 0; i < 8; ++i) { b[i] = lo[i]; b[i + 8] = hi[i]; }
  return b;
}

// C/D 16x16 f32 layout: VGPR v holds row M=v for lanes 0-15 (N=lane) and row
// M=v+8 for lanes 16-31 (N=lane-16).
__device__ __forceinline__ void store_tile(const v8f* acc, const float* bias,
                                           __bf16* out, int rowbase, int M,
                                           int lane, int relu) {
  int nl = lane & 15;
  int roff = (lane >> 4) << 3;
#pragma unroll
  for (int nt = 0; nt < 8; ++nt) {
    float bv = bias[nt * 16 + nl];
#pragma unroll
    for (int v = 0; v < 8; ++v) {
      float x = acc[nt][v] + bv;
      if (relu) x = fmaxf(x, 0.f);
      int r = rowbase + roff + v;
      if (r < M) out[(size_t)r * CDIM + nt * 16 + nl] = (__bf16)x;
    }
  }
}

// ---- weight prep ------------------------------------------------------------
__global__ __launch_bounds__(256) void k_pack_b_f32(const float* src, __bf16* dst,
                                                    int K, int N) {
  int idx = blockIdx.x * 256 + threadIdx.x;
  if (idx >= K * N) return;
  int i = idx & 15;
  int lane = (idx >> 4) & 31;
  int t2 = idx >> 9;
  int ntiles = N >> 4;
  int nt = t2 % ntiles;
  int kt = t2 / ntiles;
  int col = nt * 16 + (lane & 15);
  int k = kt * 32 + ((lane >> 4) << 3) + (i & 7) + ((i >> 3) << 4);
  dst[idx] = (__bf16)src[(size_t)k * N + col];
}

// Concatenated SAGE weight: rows 0..127 = Wl_a, 128..255 = Wl_b,
// 256..383 = Wr_a + Wr_b (x@Wr_a + x@Wr_b = x@(Wr_a+Wr_b)).
__global__ __launch_bounds__(256) void k_pack_wcat(const float* Wla, const float* Wlb,
                                                   const float* Wra, const float* Wrb,
                                                   __bf16* dst) {
  int idx = blockIdx.x * 256 + threadIdx.x;
  if (idx >= FDIM * CDIM) return;
  int i = idx & 15;
  int lane = (idx >> 4) & 31;
  int t2 = idx >> 9;
  int nt = t2 & 7;
  int kt = t2 >> 3;
  int col = nt * 16 + (lane & 15);
  int k = kt * 32 + ((lane >> 4) << 3) + (i & 7) + ((i >> 3) << 4);
  float v;
  if (k < CDIM)            v = Wla[(size_t)k * CDIM + col];
  else if (k < 2 * CDIM)   v = Wlb[(size_t)(k - CDIM) * CDIM + col];
  else                     v = Wra[(size_t)(k - 2 * CDIM) * CDIM + col] +
                               Wrb[(size_t)(k - 2 * CDIM) * CDIM + col];
  dst[idx] = (__bf16)v;
}

__global__ void k_bias2(const float* a, const float* b, float* d, int n) {
  int i = blockIdx.x * 256 + threadIdx.x;
  if (i < n) d[i] = a[i] + b[i];
}

__global__ void k_zero(float* p, int n) {
  int i = blockIdx.x * 256 + threadIdx.x;
  if (i < n) p[i] = 0.f;
}

// xp = prof_emb[professor_node_id]  (gather; id happens to be arange)
__global__ void k_gather_emb(const float* emb, const int* pid, __bf16* xp, int n) {
  int i = blockIdx.x * 256 + threadIdx.x;
  if (i >= n) return;
  int p = i / CDIM, c = i % CDIM;
  xp[i] = (__bf16)emb[(size_t)pid[p] * CDIM + c];
}

// ---- aggregations (segment_sum via f32 atomics) -----------------------------
// One 32-thread wave per row: each lane handles 4 channels.
__global__ __launch_bounds__(256) void k_agg_sup(const __bf16* xt, const int* mentor,
                                                 float* a0, int n) {
  int t = blockIdx.x * 256 + threadIdx.x;
  int i = t >> 5;
  if (i >= n) return;
  int c = (t & 31) << 2;
  int d = mentor[i];
  v4bf v = *reinterpret_cast<const v4bf*>(xt + (size_t)i * CDIM + c);
  float* dst = a0 + (size_t)d * CDIM + c;
  atomicAdd(dst + 0, (float)v[0]);
  atomicAdd(dst + 1, (float)v[1]);
  atomicAdd(dst + 2, (float)v[2]);
  atomicAdd(dst + 3, (float)v[3]);
}

// Committee edges: a1[com1] += xt[com0]   and   a3[com0] += xp[com1].
__global__ __launch_bounds__(256) void k_agg_com(const __bf16* xt, const __bf16* xp,
                                                 const int* com0, const int* com1,
                                                 float* a1, float* a3, int n) {
  int t = blockIdx.x * 256 + threadIdx.x;
  int e = t >> 5;
  if (e >= n) return;
  int c = (t & 31) << 2;
  int s = com0[e], d = com1[e];
  v4bf vt = *reinterpret_cast<const v4bf*>(xt + (size_t)s * CDIM + c);
  v4bf vp = *reinterpret_cast<const v4bf*>(xp + (size_t)d * CDIM + c);
  float* d1 = a1 + (size_t)d * CDIM + c;
  float* d3 = a3 + (size_t)s * CDIM + c;
  atomicAdd(d1 + 0, (float)vt[0]); atomicAdd(d1 + 1, (float)vt[1]);
  atomicAdd(d1 + 2, (float)vt[2]); atomicAdd(d1 + 3, (float)vt[3]);
  atomicAdd(d3 + 0, (float)vp[0]); atomicAdd(d3 + 1, (float)vp[1]);
  atomicAdd(d3 + 2, (float)vp[2]); atomicAdd(d3 + 3, (float)vp[3]);
}

// ---- GEMM kernels -----------------------------------------------------------
// All GEMMs use 2 M-tiles per wave (32 rows x 128/32 cols): 16 WMMAs per k-tile
// against 4 A-loads + 16 B-loads -> 2 WMMAs of work per B-fragment wait.

// Encoder: out[M,128] = bf16( X[M,384](f32) @ Wenc + b )
__global__ __launch_bounds__(256) void k_encode(const float* X, const __bf16* Bp,
                                                const float* bias, __bf16* out, int M) {
  int lane = threadIdx.x & 31, wv = threadIdx.x >> 5;
  int rowbase = (blockIdx.x * 8 + wv) << 5;
  if (rowbase >= M) return;
  int row0 = imin(rowbase + (lane & 15), M - 1);
  int row1 = imin(rowbase + 16 + (lane & 15), M - 1);
  v8f acc0[8] = {}, acc1[8] = {};
  for (int kt = 0; kt < FDIM / 32; ++kt) {
    v16bf a0 = load_a_f32(X, FDIM, row0, kt * 32, lane);
    v16bf a1 = load_a_f32(X, FDIM, row1, kt * 32, lane);
#pragma unroll
    for (int nt = 0; nt < 8; ++nt) {
      v16bf b = load_b_pack(Bp, kt, nt, 8, lane);
      acc0[nt] = wmma_bf16(a0, b, acc0[nt]);
      acc1[nt] = wmma_bf16(a1, b, acc1[nt]);
    }
  }
  store_tile(acc0, bias, out, rowbase, M, lane, 0);
  store_tile(acc1, bias, out, rowbase + 16, M, lane, 0);
}

// Fused hetero-SAGE update: out = [s0(gather idx0) | s1 | s2] @ Wcat + bias.
// K-chunk 0..127 -> s0, 128..255 -> s1, 256..383 -> s2.  Each source is either
// bf16 (f*=0) or f32 (f*=1, read straight from the atomic accumulators).
__global__ __launch_bounds__(256) void k_layer(const void* s0, int f0, const int* idx0,
                                               const void* s1, int f1,
                                               const void* s2, int f2,
                                               const __bf16* Bp, const float* bias,
                                               __bf16* out, int M, int relu) {
  int lane = threadIdx.x & 31, wv = threadIdx.x >> 5;
  int rowbase = (blockIdx.x * 8 + wv) << 5;
  if (rowbase >= M) return;
  int row0 = imin(rowbase + (lane & 15), M - 1);
  int row1 = imin(rowbase + 16 + (lane & 15), M - 1);
  int g0 = idx0 ? idx0[row0] : row0;
  int g1 = idx0 ? idx0[row1] : row1;
  v8f acc0[8] = {}, acc1[8] = {};
#pragma unroll
  for (int kt = 0; kt < 12; ++kt) {
    int chunk = kt >> 2;
    int kl = (kt & 3) * 32;
    const void* src = (chunk == 0) ? s0 : (chunk == 1 ? s1 : s2);
    int isF = (chunk == 0) ? f0 : (chunk == 1 ? f1 : f2);
    int ar0 = (chunk == 0) ? g0 : row0;
    int ar1 = (chunk == 0) ? g1 : row1;
    v16bf a0 = load_a_any(src, isF, CDIM, ar0, kl, lane);
    v16bf a1 = load_a_any(src, isF, CDIM, ar1, kl, lane);
#pragma unroll
    for (int nt = 0; nt < 8; ++nt) {
      v16bf b = load_b_pack(Bp, kt, nt, 8, lane);
      acc0[nt] = wmma_bf16(a0, b, acc0[nt]);
      acc1[nt] = wmma_bf16(a1, b, acc1[nt]);
    }
  }
  store_tile(acc0, bias, out, rowbase, M, lane, relu);
  store_tile(acc1, bias, out, rowbase + 16, M, lane, relu);
}

// Classifier: h = relu([xt[e0]|xp[mentor[e0]]|xp[e1]] @ W1 + b1);
// pred = h @ W2 + b2, with the final 32-dot fused as a wave shuffle reduce.
__global__ __launch_bounds__(256) void k_cls(const __bf16* xt, const __bf16* xp,
                                             const int* e0, const int* e1,
                                             const int* mentor, const __bf16* Bp,
                                             const float* b1, const float* W2,
                                             const float* b2, float* pred, int M) {
  int lane = threadIdx.x & 31, wv = threadIdx.x >> 5;
  int rowbase = (blockIdx.x * 8 + wv) << 5;
  if (rowbase >= M) return;
  int row0 = imin(rowbase + (lane & 15), M - 1);
  int row1 = imin(rowbase + 16 + (lane & 15), M - 1);
  int i0a = e0[row0], i1a = e1[row0], ima = mentor[i0a];
  int i0b = e0[row1], i1b = e1[row1], imb = mentor[i0b];
  v8f acc0[2] = {}, acc1[2] = {};
#pragma unroll
  for (int kt = 0; kt < 12; ++kt) {
    int chunk = kt >> 2;
    int kl = (kt & 3) * 32;
    const __bf16* src = (chunk == 0) ? xt : xp;
    int ar0 = (chunk == 0) ? i0a : (chunk == 1 ? ima : i1a);
    int ar1 = (chunk == 0) ? i0b : (chunk == 1 ? imb : i1b);
    v16bf a0 = load_a_bf16(src, ar0, kl, lane);
    v16bf a1 = load_a_bf16(src, ar1, kl, lane);
#pragma unroll
    for (int nt = 0; nt < 2; ++nt) {
      v16bf b = load_b_pack(Bp, kt, nt, 2, lane);
      acc0[nt] = wmma_bf16(a0, b, acc0[nt]);
      acc1[nt] = wmma_bf16(a1, b, acc1[nt]);
    }
  }
  int nl = lane & 15;
  float ba = b1[nl], bb = b1[16 + nl];
  float wa = W2[nl], wb = W2[16 + nl];
  float bias2 = b2[0];
#pragma unroll
  for (int t = 0; t < 2; ++t) {
    const v8f* A = t ? acc1 : acc0;
    int base = rowbase + t * 16;
#pragma unroll
    for (int v = 0; v < 8; ++v) {
      float h0 = fmaxf(A[0][v] + ba, 0.f) * wa;
      float h1 = fmaxf(A[1][v] + bb, 0.f) * wb;
      float s = h0 + h1;
      // reduce across each 16-lane half (rows M=v and M=v+8 independently)
      s += __shfl_xor(s, 1);
      s += __shfl_xor(s, 2);
      s += __shfl_xor(s, 4);
      s += __shfl_xor(s, 8);
      int r = base + ((lane >> 4) << 3) + v;
      if (nl == 0 && r < M) pred[r] = s + bias2;
    }
  }
}

// ---- host orchestration -----------------------------------------------------
static inline int cdiv(long a, long b) { return (int)((a + b - 1) / b); }

extern "C" void kernel_launch(void* const* d_in, const int* in_sizes, int n_in,
                              void* d_out, int out_size, void* d_ws, size_t ws_size,
                              hipStream_t stream) {
  const float* thesis_x = (const float*)d_in[0];
  const float* Wenc     = (const float*)d_in[1];
  const float* benc     = (const float*)d_in[2];
  const float* prof_emb = (const float*)d_in[3];
  const float* gWl      = (const float*)d_in[4];   // [2,4,C,C]
  const float* gbl      = (const float*)d_in[5];   // [2,4,C]
  const float* gWr      = (const float*)d_in[6];   // [2,4,C,C]
  const float* clsW1    = (const float*)d_in[7];   // [384,32]
  const float* clsb1    = (const float*)d_in[8];   // [32]
  const float* clsW2    = (const float*)d_in[9];   // [32]
  const float* clsb2    = (const float*)d_in[10];  // [1]
  const int*   pid      = (const int*)d_in[11];
  const int*   sup      = (const int*)d_in[12];    // [2,NT]
  const int*   com      = (const int*)d_in[13];    // [2,EC]
  const int*   el       = (const int*)d_in[14];    // [2,EL]
  const int* mentor = sup + NT;
  const int* com0 = com,  * com1 = com + ECNT;
  const int* el0  = el,   * el1  = el + ELAB;
  float* pred = (float*)d_out;

  char* ws = (char*)d_ws;
  size_t off = 0;
  auto alloc = [&](size_t bytes) -> char* {
    char* p = ws + off;
    off = (off + bytes + 255) & ~(size_t)255;
    return p;
  };
  __bf16* xt_a = (__bf16*)alloc((size_t)NT * CDIM * 2);
  __bf16* xt_b = (__bf16*)alloc((size_t)NT * CDIM * 2);
  __bf16* xp_a = (__bf16*)alloc((size_t)NP * CDIM * 2);
  __bf16* xp_b = (__bf16*)alloc((size_t)NP * CDIM * 2);
  float*  a0f  = (float*)alloc((size_t)NP * CDIM * 4);
  float*  a1f  = (float*)alloc((size_t)NP * CDIM * 4);
  float*  a3f  = (float*)alloc((size_t)NT * CDIM * 4);
  __bf16* pWenc = (__bf16*)alloc((size_t)FDIM * CDIM * 2);
  __bf16* pWt[2] = {(__bf16*)alloc((size_t)FDIM * CDIM * 2),
                    (__bf16*)alloc((size_t)FDIM * CDIM * 2)};
  __bf16* pWp[2] = {(__bf16*)alloc((size_t)FDIM * CDIM * 2),
                    (__bf16*)alloc((size_t)FDIM * CDIM * 2)};
  __bf16* pWcls = (__bf16*)alloc((size_t)FDIM * 32 * 2);
  float* biasT[2] = {(float*)alloc(CDIM * 4), (float*)alloc(CDIM * 4)};
  float* biasP[2] = {(float*)alloc(CDIM * 4), (float*)alloc(CDIM * 4)};

  // -- weight prep (tiny, one-time per launch) --
  k_pack_b_f32<<<cdiv(FDIM * CDIM, 256), 256, 0, stream>>>(Wenc, pWenc, FDIM, CDIM);
  k_pack_b_f32<<<cdiv(FDIM * 32, 256), 256, 0, stream>>>(clsW1, pWcls, FDIM, 32);
  const size_t WSZ = (size_t)CDIM * CDIM;
  for (int l = 0; l < 2; ++l) {
    // thesis update uses edge types 2 (rev sup -> xp[mentor]) and 3 (rev com -> a3)
    k_pack_wcat<<<cdiv(FDIM * CDIM, 256), 256, 0, stream>>>(
        gWl + (l * 4 + 2) * WSZ, gWl + (l * 4 + 3) * WSZ,
        gWr + (l * 4 + 2) * WSZ, gWr + (l * 4 + 3) * WSZ, pWt[l]);
    k_bias2<<<1, 256, 0, stream>>>(gbl + (l * 4 + 2) * CDIM,
                                   gbl + (l * 4 + 3) * CDIM, biasT[l], CDIM);
    // professor update uses edge types 0 (sup -> a0) and 1 (com -> a1)
    k_pack_wcat<<<cdiv(FDIM * CDIM, 256), 256, 0, stream>>>(
        gWl + (l * 4 + 0) * WSZ, gWl + (l * 4 + 1) * WSZ,
        gWr + (l * 4 + 0) * WSZ, gWr + (l * 4 + 1) * WSZ, pWp[l]);
    k_bias2<<<1, 256, 0, stream>>>(gbl + (l * 4 + 0) * CDIM,
                                   gbl + (l * 4 + 1) * CDIM, biasP[l], CDIM);
  }

  // -- node features --
  k_gather_emb<<<cdiv((long)NP * CDIM, 256), 256, 0, stream>>>(prof_emb, pid, xp_a,
                                                               NP * CDIM);
  k_encode<<<cdiv(cdiv(NT, 32), 8), 256, 0, stream>>>(thesis_x, pWenc, benc, xt_a, NT);

  // -- two hetero-SAGE layers (ping-pong xt_a/xp_a <-> xt_b/xp_b) --
  __bf16 *xt_cur = xt_a, *xp_cur = xp_a, *xt_nxt = xt_b, *xp_nxt = xp_b;
  for (int l = 0; l < 2; ++l) {
    k_zero<<<cdiv((long)NP * CDIM, 256), 256, 0, stream>>>(a0f, NP * CDIM);
    k_zero<<<cdiv((long)NP * CDIM, 256), 256, 0, stream>>>(a1f, NP * CDIM);
    k_zero<<<cdiv((long)NT * CDIM, 256), 256, 0, stream>>>(a3f, NT * CDIM);
    k_agg_sup<<<cdiv((long)NT * 32, 256), 256, 0, stream>>>(xt_cur, mentor, a0f, NT);
    k_agg_com<<<cdiv((long)ECNT * 32, 256), 256, 0, stream>>>(xt_cur, xp_cur, com0,
                                                              com1, a1f, a3f, ECNT);
    int relu = (l == 0) ? 1 : 0;
    // out_t = xp[mentor]@Wl2 + a3@Wl3 + xt@(Wr2+Wr3) + (b2+b3)
    k_layer<<<cdiv(cdiv(NT, 32), 8), 256, 0, stream>>>(
        xp_cur, 0, mentor, a3f, 1, xt_cur, 0, pWt[l], biasT[l], xt_nxt, NT, relu);
    // out_p = a0@Wl0 + a1@Wl1 + xp@(Wr0+Wr1) + (b0+b1)
    k_layer<<<cdiv(cdiv(NP, 32), 8), 256, 0, stream>>>(
        a0f, 1, (const int*)nullptr, a1f, 1, xp_cur, 0, pWp[l], biasP[l],
        xp_nxt, NP, relu);
    __bf16* t;
    t = xt_cur; xt_cur = xt_nxt; xt_nxt = t;
    t = xp_cur; xp_cur = xp_nxt; xp_nxt = t;
  }

  // -- classifier --
  k_cls<<<cdiv(cdiv(ELAB, 32), 8), 256, 0, stream>>>(xt_cur, xp_cur, el0, el1,
                                                     mentor, pWcls, clsb1, clsW2,
                                                     clsb2, pred, ELAB);
  (void)in_sizes; (void)n_in; (void)out_size; (void)ws_size;
}
